// EfficientSpatialWindowAttention_32040456028388
// MI455X (gfx1250) — matrix operand order repaired and checked by reading, not compiled, not run on hardware
//
#include <hip/hip_runtime.h>
#include <hip/hip_bf16.h>
#include <math.h>

#define DI __device__ __forceinline__

typedef __attribute__((ext_vector_type(16))) __bf16 bf16x16;
typedef __attribute__((ext_vector_type(8)))  float  f32x8;

union Frag { bf16x16 v; uint4 q[2]; __bf16 h[16]; };

static constexpr int Bb = 16, Nn = 1024, Cc = 768, Hh = 12, Dd = 64, BANKN = 256;
static constexpr int Mrows = Bb * Nn;        // 16384
static constexpr int KVLEN = Nn + BANKN;     // 1280
static constexpr float ATT_SCALE = 0.125f;   // 1/sqrt(64)

// workspace byte offsets
static constexpr size_t Q_BYTES  = (size_t)Bb * Hh * Nn * Dd * 2;     // 25165824
static constexpr size_t K_BYTES  = (size_t)Bb * Hh * KVLEN * Dd * 2;  // 31457280
static constexpr size_t PW_BYTES = (size_t)Cc * Cc * 2;               // 1179648
static constexpr size_t Q_OFF  = 0;
static constexpr size_t K_OFF  = Q_OFF + Q_BYTES;
static constexpr size_t V_OFF  = K_OFF + K_BYTES;
static constexpr size_t PW_OFF = V_OFF + K_BYTES;

DI f32x8 fzero() { f32x8 z = {0.f,0.f,0.f,0.f,0.f,0.f,0.f,0.f}; return z; }

DI void cvt4(__bf16* dst, float4 v) {
  dst[0] = (__bf16)v.x; dst[1] = (__bf16)v.y;
  dst[2] = (__bf16)v.z; dst[3] = (__bf16)v.w;
}

DI float rmax16(float v) {
  v = fmaxf(v, __shfl_xor(v, 1, 32));
  v = fmaxf(v, __shfl_xor(v, 2, 32));
  v = fmaxf(v, __shfl_xor(v, 4, 32));
  v = fmaxf(v, __shfl_xor(v, 8, 32));
  return v;
}
DI float rsum16(float v) {
  v += __shfl_xor(v, 1, 32);
  v += __shfl_xor(v, 2, 32);
  v += __shfl_xor(v, 4, 32);
  v += __shfl_xor(v, 8, 32);
  return v;
}

// RA = reuse-A hint (OPSEL): set on every WMMA of a clause except the last,
// since consecutive instructions are identical-opcode and share the A operand.
template <bool RA>
DI f32x8 wmma_bf16(const Frag& a, const Frag& b, f32x8 c) {
  return __builtin_amdgcn_wmma_f32_16x16x32_bf16(
      false, a.v, false, b.v, (short)0, c, RA, false);
}

// ---------------------------------------------------------------------------
// Kernel 1: QKV = x @ qkv_w^T + qkv_b, scattered to head-major bf16 q/k/v.
// Grid: (36 col tiles of 64, 256 row tiles of 64). 128 threads = 4 waves.
// Double-buffered LDS, one barrier per k-step, global loads overlap WMMA.
// ---------------------------------------------------------------------------
__global__ __launch_bounds__(128)
void k_qkv_gemm(const float* __restrict__ x, const float* __restrict__ w,
                const float* __restrict__ bias,
                __bf16* __restrict__ qO, __bf16* __restrict__ kO,
                __bf16* __restrict__ vO) {
  __shared__ __align__(16) __bf16 sA[2][64 * 32];
  __shared__ __align__(16) __bf16 sB[2][64 * 32];
  const int nb = blockIdx.x, mb = blockIdx.y;
  const int tid = threadIdx.x, lane = tid & 31, wv = tid >> 5;
  const int ln = lane & 15, hi = lane >> 4;
  const int gm0 = mb * 64, gn0 = nb * 64;

  f32x8 acc[4];
#pragma unroll
  for (int t = 0; t < 4; ++t) acc[t] = fzero();

  float4 rA[4], rB[4];  // staged tile for next k-step (512 float4 / 128 thr)
#pragma unroll
  for (int i = 0; i < 4; ++i) {
    const int slot = tid + i * 128;
    const int r = slot >> 3, c4 = slot & 7;
    rA[i] = *(const float4*)&x[(size_t)(gm0 + r) * Cc + c4 * 4];
    rB[i] = *(const float4*)&w[(size_t)(gn0 + r) * Cc + c4 * 4];
  }

  for (int kb = 0; kb < 24; ++kb) {
    const int buf = kb & 1;
#pragma unroll
    for (int i = 0; i < 4; ++i) {
      const int slot = tid + i * 128;
      const int r = slot >> 3, c4 = slot & 7;
      union { __bf16 h[4]; uint2 u; } pa, pb;
      cvt4(pa.h, rA[i]);
      cvt4(pb.h, rB[i]);
      *(uint2*)&sA[buf][r * 32 + c4 * 4] = pa.u;
      *(uint2*)&sB[buf][r * 32 + c4 * 4] = pb.u;
    }
    __syncthreads();
    if (kb < 23) {
#pragma unroll
      for (int i = 0; i < 4; ++i) {
        const int slot = tid + i * 128;
        const int r = slot >> 3, c4 = slot & 7;
        rA[i] = *(const float4*)&x[(size_t)(gm0 + r) * Cc + (kb + 1) * 32 + c4 * 4];
        rB[i] = *(const float4*)&w[(size_t)(gn0 + r) * Cc + (kb + 1) * 32 + c4 * 4];
      }
    }
    Frag a;
    const __bf16* arow = sA[buf] + (wv * 16 + ln) * 32;
    a.q[0] = *(const uint4*)(arow + hi * 16);
    a.q[1] = *(const uint4*)(arow + hi * 16 + 8);
    Frag bfr[4];
#pragma unroll
    for (int t = 0; t < 4; ++t) {
      const __bf16* brow = sB[buf] + (t * 16 + ln) * 32;
      bfr[t].q[0] = *(const uint4*)(brow + hi * 16);
      bfr[t].q[1] = *(const uint4*)(brow + hi * 16 + 8);
    }
#pragma unroll
    for (int t = 0; t < 3; ++t) acc[t] = wmma_bf16<true>(a, bfr[t], acc[t]);
    acc[3] = wmma_bf16<false>(a, bfr[3], acc[3]);
  }

  // 64-col block lies in exactly one (q|k|v, head) slice: 64 | 768.
  const int t3 = nb / 12, h = nb % 12;
  __bf16* dbase;
  int L;
  if (t3 == 0)      { dbase = qO; L = Nn; }
  else if (t3 == 1) { dbase = kO; L = KVLEN; }
  else              { dbase = vO; L = KVLEN; }
  dbase += (size_t)h * L * Dd;
  const size_t bstride = (size_t)12 * L * Dd;
#pragma unroll
  for (int t = 0; t < 4; ++t) {
    const int d = t * 16 + ln;
    const float bv = bias[gn0 + t * 16 + ln];
#pragma unroll
    for (int r = 0; r < 8; ++r) {
      const int gm = gm0 + wv * 16 + r + 8 * hi;
      const int b = gm >> 10, n = gm & 1023;
      dbase[(size_t)b * bstride + (size_t)n * Dd + d] = (__bf16)(acc[t][r] + bv);
    }
  }
}

// ---------------------------------------------------------------------------
// Kernel 2: broadcast bank_k/bank_v into rows 1024..1279 of k/v_full.
// ---------------------------------------------------------------------------
__global__ __launch_bounds__(256)
void k_bank_fill(const float* __restrict__ bk, const float* __restrict__ bv,
                 __bf16* __restrict__ kO, __bf16* __restrict__ vO) {
  const int e = blockIdx.x * 256 + threadIdx.x;  // < 196608
  const int h = e >> 14;
  const int m = (e >> 6) & 255;
  const int d = e & 63;
  const __bf16 kb16 = (__bf16)bk[m * Cc + h * 64 + d];
  const __bf16 vb16 = (__bf16)bv[m * Cc + h * 64 + d];
#pragma unroll
  for (int b = 0; b < 16; ++b) {
    const size_t off = ((size_t)(b * 12 + h) * KVLEN + Nn + m) * Dd + d;
    kO[off] = kb16;
    vO[off] = vb16;
  }
}

// ---------------------------------------------------------------------------
// Kernel 2b: one-shot f32 -> bf16 weight conversion (proj_w).
// ---------------------------------------------------------------------------
__global__ __launch_bounds__(256)
void k_cvt_w(const float* __restrict__ src, __bf16* __restrict__ dst) {
  const size_t i = ((size_t)blockIdx.x * 256 + threadIdx.x) * 4;
  float4 v = *(const float4*)(src + i);
  union { __bf16 h[4]; uint2 u; } pk;
  cvt4(pk.h, v);
  *(uint2*)(dst + i) = pk.u;
}

// ---------------------------------------------------------------------------
// Kernel 3: flash-style attention. Grid (16 q-blocks, 192 bh). 128 threads.
// Double-buffered K/V LDS; V transposed at stage time so P@V B-operands are
// contiguous ds_load_b128.
// ---------------------------------------------------------------------------
__global__ __launch_bounds__(128)
void k_attn(const __bf16* __restrict__ qB, const __bf16* __restrict__ kB,
            const __bf16* __restrict__ vB, float* __restrict__ out) {
  __shared__ __align__(16) __bf16 sK[2][64 * 64];
  __shared__ __align__(16) __bf16 sVt[2][64 * 64];
  __shared__ __align__(16) __bf16 sP[4][16 * 64];
  const int qbk = blockIdx.x, bh = blockIdx.y;
  const int b = bh / 12, h = bh % 12;
  const int tid = threadIdx.x, lane = tid & 31, wv = tid >> 5;
  const int ln = lane & 15, hi = lane >> 4;

  Frag aq[2];
  {
    const __bf16* qrow =
        qB + ((size_t)(b * 12 + h) * Nn + qbk * 64 + wv * 16 + ln) * Dd;
#pragma unroll
    for (int s = 0; s < 2; ++s) {
      aq[s].q[0] = *(const uint4*)(qrow + s * 32 + hi * 16);
      aq[s].q[1] = *(const uint4*)(qrow + s * 32 + hi * 16 + 8);
    }
  }

  f32x8 O[4];
  float m[8], l[8];
#pragma unroll
  for (int t = 0; t < 4; ++t) O[t] = fzero();
#pragma unroll
  for (int r = 0; r < 8; ++r) { m[r] = -INFINITY; l[r] = 0.f; }

  const __bf16* kbase = kB + (size_t)(b * 12 + h) * KVLEN * Dd;
  const __bf16* vbase = vB + (size_t)(b * 12 + h) * KVLEN * Dd;

  uint4 kR[4];
  unsigned vR[16];
  {
    const uint4* ks = (const uint4*)kbase;
    const unsigned* vs = (const unsigned*)vbase;
#pragma unroll
    for (int i = 0; i < 4; ++i) kR[i] = ks[tid + i * 128];
#pragma unroll
    for (int i = 0; i < 16; ++i) vR[i] = vs[tid + i * 128];
  }

  for (int kb = 0; kb < 20; ++kb) {
    const int buf = kb & 1;
    {
      uint4* kd = (uint4*)sK[buf];
#pragma unroll
      for (int i = 0; i < 4; ++i) kd[tid + i * 128] = kR[i];
#pragma unroll
      for (int i = 0; i < 16; ++i) {
        const int idx = tid + i * 128;
        const int key = idx >> 5, dp = idx & 31;
        union { unsigned u; __bf16 hh[2]; } cv;
        cv.u = vR[i];
        sVt[buf][(2 * dp) * 64 + key] = cv.hh[0];
        sVt[buf][(2 * dp + 1) * 64 + key] = cv.hh[1];
      }
    }
    __syncthreads();
    if (kb < 19) {
      const uint4* ks = (const uint4*)(kbase + (size_t)(kb + 1) * 64 * 64);
      const unsigned* vs = (const unsigned*)(vbase + (size_t)(kb + 1) * 64 * 64);
#pragma unroll
      for (int i = 0; i < 4; ++i) kR[i] = ks[tid + i * 128];
#pragma unroll
      for (int i = 0; i < 16; ++i) vR[i] = vs[tid + i * 128];
    }

    // S = q @ k^T
    f32x8 S[4];
#pragma unroll
    for (int t = 0; t < 4; ++t) S[t] = fzero();
#pragma unroll
    for (int s = 0; s < 2; ++s) {
      Frag bk_[4];
#pragma unroll
      for (int t = 0; t < 4; ++t) {
        const __bf16* row = sK[buf] + (t * 16 + ln) * 64 + s * 32 + hi * 16;
        bk_[t].q[0] = *(const uint4*)(row);
        bk_[t].q[1] = *(const uint4*)(row + 8);
      }
#pragma unroll
      for (int t = 0; t < 3; ++t) S[t] = wmma_bf16<true>(aq[s], bk_[t], S[t]);
      S[3] = wmma_bf16<false>(aq[s], bk_[3], S[3]);
    }
#pragma unroll
    for (int t = 0; t < 4; ++t)
#pragma unroll
      for (int r = 0; r < 8; ++r) S[t][r] *= ATT_SCALE;

    // online softmax
    float sc[8], rs[8];
#pragma unroll
    for (int r = 0; r < 8; ++r) {
      float pm = fmaxf(fmaxf(S[0][r], S[1][r]), fmaxf(S[2][r], S[3][r]));
      pm = rmax16(pm);
      const float mn = fmaxf(m[r], pm);
      sc[r] = __expf(m[r] - mn);
      m[r] = mn;
      rs[r] = 0.f;
    }
#pragma unroll
    for (int t = 0; t < 4; ++t) {
#pragma unroll
      for (int r = 0; r < 8; ++r) {
        const float e = __expf(S[t][r] - m[r]);
        rs[r] += e;
        sP[wv][(r + 8 * hi) * 64 + t * 16 + ln] = (__bf16)e;
      }
    }
#pragma unroll
    for (int r = 0; r < 8; ++r) {
      l[r] = l[r] * sc[r] + rsum16(rs[r]);
#pragma unroll
      for (int td = 0; td < 4; ++td) O[td][r] *= sc[r];
    }

    // O += P @ V
#pragma unroll
    for (int s = 0; s < 2; ++s) {
      Frag ap;
      const __bf16* prow = sP[wv] + ln * 64 + s * 32;
      ap.q[0] = *(const uint4*)(prow + hi * 16);
      ap.q[1] = *(const uint4*)(prow + hi * 16 + 8);
      Frag bv_[4];
#pragma unroll
      for (int td = 0; td < 4; ++td) {
        const __bf16* vrow = sVt[buf] + (td * 16 + ln) * 64 + s * 32 + hi * 16;
        bv_[td].q[0] = *(const uint4*)(vrow);
        bv_[td].q[1] = *(const uint4*)(vrow + 8);
      }
#pragma unroll
      for (int td = 0; td < 3; ++td) O[td] = wmma_bf16<true>(ap, bv_[td], O[td]);
      O[3] = wmma_bf16<false>(ap, bv_[3], O[3]);
    }
  }

#pragma unroll
  for (int r = 0; r < 8; ++r) {
    const float inv = 1.0f / l[r];
    const int n = qbk * 64 + wv * 16 + r + 8 * hi;
    float* orow = out + ((size_t)b * Nn + n) * Cc + h * 64;
#pragma unroll
    for (int td = 0; td < 4; ++td) orow[td * 16 + ln] = O[td][r] * inv;
  }
}

// ---------------------------------------------------------------------------
// Kernel 4: p = out @ proj_w^T + proj_b, then LayerNorm over 768, fused.
// Each WG: 16 rows x all 768 cols (8 waves x 96 cols). 1024 blocks.
// A tile staged once per k-step into double-buffered LDS (shared by all 8
// waves); B fragments loaded directly from pre-converted bf16 proj_w.
// ---------------------------------------------------------------------------
__global__ __launch_bounds__(256)
void k_proj_ln(const float* __restrict__ outF, const __bf16* __restrict__ pwb,
               const float* __restrict__ pb, const float* __restrict__ nw,
               const float* __restrict__ nbias, float* __restrict__ dst) {
  __shared__ __align__(16) __bf16 sA2[2][16 * 32];
  __shared__ float rsum[16], rsq[16];
  const int blk = blockIdx.x;
  const int tid = threadIdx.x, lane = tid & 31, wv = tid >> 5;
  const int ln = lane & 15, hi = lane >> 4;

  f32x8 acc[6];
#pragma unroll
  for (int t = 0; t < 6; ++t) acc[t] = fzero();

  // A staging: 512 elements, 2 per thread
  const int ar_r = tid >> 4;            // row 0..15
  const int ar_c = (tid & 15) * 2;      // even col 0..30
  const float* aptr = outF + ((size_t)blk * 16 + ar_r) * Cc + ar_c;
  float2 rA = *(const float2*)(aptr);

  for (int kb = 0; kb < 24; ++kb) {
    const int buf = kb & 1;
    {
      union { __bf16 h[2]; unsigned u; } pk;
      pk.h[0] = (__bf16)rA.x;
      pk.h[1] = (__bf16)rA.y;
      *(unsigned*)&sA2[buf][ar_r * 32 + ar_c] = pk.u;
    }
    __syncthreads();
    if (kb < 23) rA = *(const float2*)(aptr + (kb + 1) * 32);

    Frag a;
    const __bf16* ar = sA2[buf] + ln * 32;
    a.q[0] = *(const uint4*)(ar + hi * 16);
    a.q[1] = *(const uint4*)(ar + hi * 16 + 8);
    Frag bf[6];
#pragma unroll
    for (int t = 0; t < 6; ++t) {
      const __bf16* bp = pwb + (size_t)(wv * 96 + t * 16 + ln) * Cc + kb * 32 + hi * 16;
      bf[t].q[0] = *(const uint4*)(bp);
      bf[t].q[1] = *(const uint4*)(bp + 8);
    }
#pragma unroll
    for (int t = 0; t < 5; ++t) acc[t] = wmma_bf16<true>(a, bf[t], acc[t]);
    acc[5] = wmma_bf16<false>(a, bf[5], acc[5]);
  }

  float nwv[6], nbv[6];
#pragma unroll
  for (int t = 0; t < 6; ++t) {
    const int o = wv * 96 + t * 16 + ln;
    const float bias = pb[o];
    nwv[t] = nw[o];
    nbv[t] = nbias[o];
#pragma unroll
    for (int r = 0; r < 8; ++r) acc[t][r] += bias;
  }

  if (tid < 16) { rsum[tid] = 0.f; rsq[tid] = 0.f; }
  __syncthreads();
#pragma unroll
  for (int r = 0; r < 8; ++r) {
    float ps = 0.f, pq = 0.f;
#pragma unroll
    for (int t = 0; t < 6; ++t) { const float v = acc[t][r]; ps += v; pq += v * v; }
    ps = rsum16(ps);
    pq = rsum16(pq);
    if (ln == 0) {
      atomicAdd(&rsum[r + 8 * hi], ps);
      atomicAdd(&rsq[r + 8 * hi], pq);
    }
  }
  __syncthreads();
#pragma unroll
  for (int r = 0; r < 8; ++r) {
    const int M = r + 8 * hi;
    const float mu = rsum[M] * (1.f / 768.f);
    const float var = rsq[M] * (1.f / 768.f) - mu * mu;
    const float rstd = rsqrtf(var + 1e-5f);
    float* drow = dst + ((size_t)blk * 16 + M) * Cc;
#pragma unroll
    for (int t = 0; t < 6; ++t) {
      const int o = wv * 96 + t * 16 + ln;
      drow[o] = (acc[t][r] - mu) * rstd * nwv[t] + nbv[t];
    }
  }
}

// ---------------------------------------------------------------------------
extern "C" void kernel_launch(void* const* d_in, const int* in_sizes, int n_in,
                              void* d_out, int out_size, void* d_ws,
                              size_t ws_size, hipStream_t stream) {
  const float* x      = (const float*)d_in[0];
  const float* qkv_w  = (const float*)d_in[1];
  const float* qkv_b  = (const float*)d_in[2];
  const float* proj_w = (const float*)d_in[3];
  const float* proj_b = (const float*)d_in[4];
  const float* norm_w = (const float*)d_in[5];
  const float* norm_b = (const float*)d_in[6];
  const float* bank_k = (const float*)d_in[7];
  const float* bank_v = (const float*)d_in[8];

  float* out      = (float*)d_out;
  float* bank_upd = out + (size_t)Mrows * Cc;

  __bf16* qbf = (__bf16*)((char*)d_ws + Q_OFF);
  __bf16* kbf = (__bf16*)((char*)d_ws + K_OFF);
  __bf16* vbf = (__bf16*)((char*)d_ws + V_OFF);
  __bf16* pwb = (__bf16*)((char*)d_ws + PW_OFF);

  k_qkv_gemm<<<dim3(36, 256), 128, 0, stream>>>(x, qkv_w, qkv_b, qbf, kbf, vbf);
  k_bank_fill<<<768, 256, 0, stream>>>(bank_k, bank_v, kbf, vbf);
  k_cvt_w<<<576, 256, 0, stream>>>(proj_w, pwb);
  k_attn<<<dim3(16, 192), 128, 0, stream>>>(qbf, kbf, vbf, out);
  k_proj_ln<<<1024, 256, 0, stream>>>(out, pwb, proj_b, norm_w, norm_b,
                                      bank_upd);
}
